// SelfAttentionLayer_pre_37649683317506
// MI455X (gfx1250) — matrix operand order repaired and checked by reading, not compiled, not run on hardware
//
#include <hip/hip_runtime.h>

// ---------------------------------------------------------------------------
// Problem sizes (fixed by the reference)
// ---------------------------------------------------------------------------
#define NROWS 8192
#define DIM   2048
#define DH    1024          // D/2
#define BAND  512           // energy rows processed per band
#define NBAND (NROWS / BAND)

// ---------------------------------------------------------------------------
// Vector types for WMMA fragments
// ---------------------------------------------------------------------------
typedef __bf16 v16bf __attribute__((ext_vector_type(16)));
typedef __bf16 v8bf  __attribute__((ext_vector_type(8)));   // 16-byte chunk
typedef float  v8f   __attribute__((ext_vector_type(8)));

union Frag16 { v16bf f; v8bf h[2]; };

// ---------------------------------------------------------------------------
// Async global->LDS support (gfx1250): guarded so the build can't regress.
// Builtin signature (from compiler diagnostics): first param is a v4i
// (vector_size(16) int) pointer in the global address space, second the LDS
// destination; both non-const.
// ---------------------------------------------------------------------------
#if defined(__has_builtin)
#  if __has_builtin(__builtin_amdgcn_global_load_async_to_lds_b128)
#    define USE_ASYNC_LDS 1
#  endif
#endif
#ifndef USE_ASYNC_LDS
#  define USE_ASYNC_LDS 0
#endif

#if USE_ASYNC_LDS
typedef int v4i __attribute__((vector_size(16)));
typedef v4i __attribute__((address_space(1))) gv4i;   // global
typedef v4i __attribute__((address_space(3))) lv4i;   // LDS

template <int N>
__device__ __forceinline__ void wait_asynccnt() {
#if __has_builtin(__builtin_amdgcn_s_wait_asynccnt)
  __builtin_amdgcn_s_wait_asynccnt(N);
#else
  if constexpr (N == 0)
    asm volatile("s_wait_asynccnt 0x0" ::: "memory");
  else
    asm volatile("s_wait_asynccnt 0xc" ::: "memory");
#endif
}
#endif

// ---------------------------------------------------------------------------
// Workspace layout (bytes)
// ---------------------------------------------------------------------------
#define OFF_XB 0u                       // x  bf16  [8192][2048]   33554432
#define OFF_QB 33554432u                // q  bf16  [8192][1024]   16777216
#define OFF_KB 50331648u                // k  bf16  [8192][1024]   16777216
#define OFF_VB 67108864u                // v  bf16  [8192][2048]   33554432
#define OFF_WT 100663296u               // Wt bf16  [<=2048][2048]  8388608 (reused)
#define OFF_EB 109051904u               // energy band f32 [512][8192] 16777216 (reused)
#define OFF_MR 125829120u               // row max  f32 [512]
#define OFF_SR 125831168u               // row scale f32 [512]
#define OFF_CJ 125833216u               // c  f32 [8192]
#define OFF_VC 125865984u               // c@v accum f32 [2048]
#define OFF_XM 125874176u               // colsum(x) f32 [2048]

// ---------------------------------------------------------------------------
// fp32 -> bf16 cast (grid-stride)
// ---------------------------------------------------------------------------
__global__ void cast_bf16_kernel(const float* __restrict__ s,
                                 __bf16* __restrict__ d, int n) {
  for (int i = blockIdx.x * blockDim.x + threadIdx.x; i < n;
       i += gridDim.x * blockDim.x)
    d[i] = (__bf16)s[i];
}

// ---------------------------------------------------------------------------
// Transpose + cast: W[R][C] f32 -> Wt[C][R] bf16   (block 32x8, LDS tile)
// ---------------------------------------------------------------------------
__global__ void transpose_cast_kernel(const float* __restrict__ W,
                                      __bf16* __restrict__ Wt,
                                      int R, int C) {
  __shared__ float t[32][33];
  const int c0 = blockIdx.x * 32, r0 = blockIdx.y * 32;
  const int tx = threadIdx.x, ty = threadIdx.y;
#pragma unroll
  for (int i = 0; i < 32; i += 8)
    t[ty + i][tx] = W[(size_t)(r0 + ty + i) * C + (c0 + tx)];
  __syncthreads();
#pragma unroll
  for (int i = 0; i < 32; i += 8)
    Wt[(size_t)(c0 + ty + i) * R + (r0 + tx)] = (__bf16)t[tx][ty + i];
}

// ---------------------------------------------------------------------------
// bf16 NT GEMM with WMMA:  C[M,N] = A[M,K] * B[N,K]^T
//   EPI==0 : C -> f32 (energy)
//   EPI==1 : C = relu(C + bias[col]) -> bf16 (projections)
// Block: 256 threads = 8 waves (2 x 4), macro tile 128 x 256, KC = 64.
// Double-buffered LDS; next chunk staged with async global->LDS DMA while the
// current chunk's 32 WMMAs execute.
// ---------------------------------------------------------------------------
#define BM   128
#define BN   256
#define KC   64
#define LSTR 72   // LDS row stride in elements (144 B: 16B-aligned, bank-spread)
#define ASZ  (BM * LSTR)
#define BSZ  (BN * LSTR)

template <int EPI>
__global__ __launch_bounds__(256) void gemm_nt_wmma(
    const __bf16* __restrict__ A, const __bf16* __restrict__ B,
    const float* __restrict__ bias, float* __restrict__ Cf,
    __bf16* __restrict__ Cb, int K, int lda, int ldb, int ldc) {
  __shared__ __bf16 As[2 * ASZ];   // 2 x 18432 B
  __shared__ __bf16 Bs[2 * BSZ];   // 2 x 36864 B

  const int tid    = threadIdx.x;
  const int lane   = tid & 31;
  const int wid    = tid >> 5;
  const int half   = lane >> 4;      // 0 | 1
  const int lm     = lane & 15;
  const int wm     = (wid & 1) * 64; // wave M offset inside block tile
  const int wn     = (wid >> 1) * 64;
  const int blockM = blockIdx.y * BM;
  const int blockN = blockIdx.x * BN;

  // global->LDS copy mapping (per thread: 4 x b128 of A, 8 x b128 of B)
  const int aRow = tid & (BM - 1);
  const int aSeg = (tid >> 7) * 32;          // 0 or 32 (elements)
  const __bf16* aG = A + (size_t)(blockM + aRow) * lda + aSeg;
  const __bf16* bG = B + (size_t)(blockN + tid) * ldb;

  auto copy_chunk = [&](int k0, int buf) {
    __bf16* aS = As + buf * ASZ + aRow * LSTR + aSeg;
    __bf16* bS = Bs + buf * BSZ + tid * LSTR;
#if USE_ASYNC_LDS
#pragma unroll
    for (int i = 0; i < 4; ++i)
      __builtin_amdgcn_global_load_async_to_lds_b128(
          (gv4i*)(aG + k0 + i * 8), (lv4i*)(aS + i * 8), 0, 0);
#pragma unroll
    for (int i = 0; i < 8; ++i)
      __builtin_amdgcn_global_load_async_to_lds_b128(
          (gv4i*)(bG + k0 + i * 8), (lv4i*)(bS + i * 8), 0, 0);
#else
#pragma unroll
    for (int i = 0; i < 4; ++i)
      *(v8bf*)(aS + i * 8) = *(const v8bf*)(aG + k0 + i * 8);
#pragma unroll
    for (int i = 0; i < 8; ++i)
      *(v8bf*)(bS + i * 8) = *(const v8bf*)(bG + k0 + i * 8);
#endif
  };

  v8f acc[4][4];
  const v8f z8 = {0.f, 0.f, 0.f, 0.f, 0.f, 0.f, 0.f, 0.f};
#pragma unroll
  for (int i = 0; i < 4; ++i)
#pragma unroll
    for (int j = 0; j < 4; ++j) acc[i][j] = z8;

  // prologue: stage chunk 0 into buffer 0
  copy_chunk(0, 0);

  int cur = 0;
  for (int k0 = 0; k0 < K; k0 += KC) {
    const bool more = (k0 + KC) < K;
    if (more) copy_chunk(k0 + KC, cur ^ 1);   // overlap next DMA with compute
#if USE_ASYNC_LDS
    if (more)
      wait_asynccnt<12>();   // chunk k's 12 per-wave async ops have landed
    else
      wait_asynccnt<0>();
#endif
    __syncthreads();

    const __bf16* Ac = As + cur * ASZ;
    const __bf16* Bc = Bs + cur * BSZ;
#pragma unroll
    for (int ks = 0; ks < KC; ks += 32) {
      Frag16 bfr[4];
#pragma unroll
      for (int nt = 0; nt < 4; ++nt) {
        const __bf16* p = Bc + (wn + nt * 16 + lm) * LSTR + ks + half * 16;
        bfr[nt].h[0] = *(const v8bf*)(p);
        bfr[nt].h[1] = *(const v8bf*)(p + 8);
      }
#pragma unroll
      for (int mt = 0; mt < 4; ++mt) {
        Frag16 afr;
        const __bf16* p = Ac + (wm + mt * 16 + lm) * LSTR + ks + half * 16;
        afr.h[0] = *(const v8bf*)(p);
        afr.h[1] = *(const v8bf*)(p + 8);
#pragma unroll
        for (int nt = 0; nt < 4; ++nt) {
          acc[mt][nt] = __builtin_amdgcn_wmma_f32_16x16x32_bf16(
              false, afr.f, false, bfr[nt].f, (short)0, acc[mt][nt], false,
              false);
        }
      }
    }
    __syncthreads();   // everyone done reading buffer `cur` before reuse
    cur ^= 1;
  }

  // ---- epilogue ----
#pragma unroll
  for (int mt = 0; mt < 4; ++mt) {
#pragma unroll
    for (int nt = 0; nt < 4; ++nt) {
      const int row0 = blockM + wm + mt * 16 + half * 8;
      const int col  = blockN + wn + nt * 16 + lm;
      float bv = 0.f;
      if constexpr (EPI == 1) bv = bias[col];
#pragma unroll
      for (int r = 0; r < 8; ++r) {
        float val = acc[mt][nt][r];
        if constexpr (EPI == 1) {
          val = fmaxf(val + bv, 0.f);
          Cb[(size_t)(row0 + r) * ldc + col] = (__bf16)val;
        } else {
          Cf[(size_t)(row0 + r) * ldc + col] = val;
        }
      }
    }
  }
}

// ---------------------------------------------------------------------------
// Row softmax stats: one wave per row. mrow = max, srow = 1/(N * sum exp)
// ---------------------------------------------------------------------------
__global__ __launch_bounds__(256) void rowstats_kernel(
    const float* __restrict__ E, float* __restrict__ mrow,
    float* __restrict__ srow, int cols) {
  const int lane = threadIdx.x & 31;
  const int row  = blockIdx.x * 8 + (threadIdx.x >> 5);
  const float* e = E + (size_t)row * cols;
  float m = -3.0e30f;
  for (int j = lane; j < cols; j += 32) m = fmaxf(m, e[j]);
#pragma unroll
  for (int o = 16; o; o >>= 1) m = fmaxf(m, __shfl_xor(m, o, 32));
  float s = 0.f;
  for (int j = lane; j < cols; j += 32) s += __expf(e[j] - m);
#pragma unroll
  for (int o = 16; o; o >>= 1) s += __shfl_xor(s, o, 32);
  if (lane == 0) {
    mrow[row] = m;
    srow[row] = 1.0f / (s * (float)NROWS);
  }
}

// ---------------------------------------------------------------------------
// c_j += sum_i exp(E[i][j]-m[i]) * scale[i]   (one column per thread)
// ---------------------------------------------------------------------------
__global__ __launch_bounds__(256) void colaccum_kernel(
    const float* __restrict__ E, const float* __restrict__ mrow,
    const float* __restrict__ srow, float* __restrict__ c, int rows,
    int cols) {
  __shared__ float ms[BAND];
  __shared__ float ss[BAND];
  for (int i = threadIdx.x; i < rows; i += blockDim.x) {
    ms[i] = mrow[i];
    ss[i] = srow[i];
  }
  __syncthreads();
  const int col = blockIdx.x * blockDim.x + threadIdx.x;
  float acc = 0.f;
  for (int i = 0; i < rows; ++i)
    acc += __expf(E[(size_t)i * cols + col] - ms[i]) * ss[i];
  atomicAdd(&c[col], acc);
}

// ---------------------------------------------------------------------------
// xm[d] += sum over a 256-row slab of X[i][d]   (f32 input)
// ---------------------------------------------------------------------------
__global__ __launch_bounds__(256) void xcolsum_kernel(
    const float* __restrict__ X, float* __restrict__ xm, int cols) {
  const int i0 = blockIdx.x * 256;
  for (int d = threadIdx.x; d < cols; d += blockDim.x) {
    float a = 0.f;
    for (int i = 0; i < 256; ++i) a += X[(size_t)(i0 + i) * cols + d];
    atomicAdd(&xm[d], a);
  }
}

// ---------------------------------------------------------------------------
// vc[d] += sum over a 256-row slab of c[j] * V[j][d]   (bf16 V)
// ---------------------------------------------------------------------------
__global__ __launch_bounds__(256) void vcolsum_kernel(
    const __bf16* __restrict__ V, const float* __restrict__ c,
    float* __restrict__ vc, int cols) {
  __shared__ float cs[256];
  const int j0 = blockIdx.x * 256;
  cs[threadIdx.x] = c[j0 + threadIdx.x];
  __syncthreads();
  for (int d = threadIdx.x; d < cols; d += blockDim.x) {
    float a = 0.f;
    for (int j = 0; j < 256; ++j)
      a += cs[j] * (float)V[(size_t)(j0 + j) * cols + d];
    atomicAdd(&vc[d], a);
  }
}

// ---------------------------------------------------------------------------
// out[d] = gamma * vc[d] + xm[d] / N
// ---------------------------------------------------------------------------
__global__ void combine_kernel(const float* __restrict__ vc,
                               const float* __restrict__ xm,
                               const float* __restrict__ gamma,
                               float* __restrict__ out, int cols) {
  const int d = blockIdx.x * blockDim.x + threadIdx.x;
  if (d < cols) out[d] = gamma[0] * vc[d] + xm[d] * (1.0f / (float)NROWS);
}

__global__ void zerof_kernel(float* __restrict__ p, int n) {
  const int i = blockIdx.x * blockDim.x + threadIdx.x;
  if (i < n) p[i] = 0.f;
}

// ---------------------------------------------------------------------------
// Host-side orchestration
// ---------------------------------------------------------------------------
extern "C" void kernel_launch(void* const* d_in, const int* in_sizes, int n_in,
                              void* d_out, int out_size, void* d_ws,
                              size_t ws_size, hipStream_t stream) {
  const float* x     = (const float*)d_in[0];
  const float* Wq    = (const float*)d_in[1];
  const float* bq    = (const float*)d_in[2];
  const float* Wk    = (const float*)d_in[3];
  const float* bk    = (const float*)d_in[4];
  const float* Wv    = (const float*)d_in[5];
  const float* bv    = (const float*)d_in[6];
  const float* gamma = (const float*)d_in[7];
  float* out = (float*)d_out;

  char* ws = (char*)d_ws;
  __bf16* xb = (__bf16*)(ws + OFF_XB);
  __bf16* qb = (__bf16*)(ws + OFF_QB);
  __bf16* kb = (__bf16*)(ws + OFF_KB);
  __bf16* vb = (__bf16*)(ws + OFF_VB);
  __bf16* wt = (__bf16*)(ws + OFF_WT);
  float*  eb = (float*)(ws + OFF_EB);
  float*  mr = (float*)(ws + OFF_MR);
  float*  sr = (float*)(ws + OFF_SR);
  float*  cj = (float*)(ws + OFF_CJ);
  float*  vc = (float*)(ws + OFF_VC);
  float*  xm = (float*)(ws + OFF_XM);

  // zero the atomic accumulators (cj, vc, xm are contiguous: 12288 floats)
  zerof_kernel<<<48, 256, 0, stream>>>(cj, 12288);

  // x -> bf16
  cast_bf16_kernel<<<4096, 256, 0, stream>>>(x, xb, NROWS * DIM);

  // ---- projections: y = relu(x @ W + b), bf16 output ----
  // q
  transpose_cast_kernel<<<dim3(DH / 32, DIM / 32), dim3(32, 8), 0, stream>>>(
      Wq, wt, DIM, DH);
  gemm_nt_wmma<1><<<dim3(DH / BN, NROWS / BM), 256, 0, stream>>>(
      xb, wt, bq, nullptr, qb, DIM, DIM, DIM, DH);
  // k
  transpose_cast_kernel<<<dim3(DH / 32, DIM / 32), dim3(32, 8), 0, stream>>>(
      Wk, wt, DIM, DH);
  gemm_nt_wmma<1><<<dim3(DH / BN, NROWS / BM), 256, 0, stream>>>(
      xb, wt, bk, nullptr, kb, DIM, DIM, DIM, DH);
  // v
  transpose_cast_kernel<<<dim3(DIM / 32, DIM / 32), dim3(32, 8), 0, stream>>>(
      Wv, wt, DIM, DIM);
  gemm_nt_wmma<1><<<dim3(DIM / BN, NROWS / BM), 256, 0, stream>>>(
      xb, wt, bv, nullptr, vb, DIM, DIM, DIM, DIM);

  // ---- energy bands: e = q k^T, softmax row stats, column accumulation ----
  for (int band = 0; band < NBAND; ++band) {
    const __bf16* qband = qb + (size_t)band * BAND * DH;
    gemm_nt_wmma<0><<<dim3(NROWS / BN, BAND / BM), 256, 0, stream>>>(
        qband, kb, nullptr, eb, nullptr, DH, DH, DH, NROWS);
    rowstats_kernel<<<BAND / 8, 256, 0, stream>>>(eb, mr, sr, NROWS);
    colaccum_kernel<<<NROWS / 256, 256, 0, stream>>>(eb, mr, sr, cj, BAND,
                                                     NROWS);
  }

  // ---- final: out = gamma * (c @ v) + mean(x, axis=0) ----
  xcolsum_kernel<<<NROWS / 256, 256, 0, stream>>>(x, xm, DIM);
  vcolsum_kernel<<<NROWS / 256, 256, 0, stream>>>(vb, cj, vc, DIM);
  combine_kernel<<<(DIM + 255) / 256, 256, 0, stream>>>(vc, xm, gamma, out,
                                                        DIM);
}
